// Attention_76338748719569
// MI455X (gfx1250) — matrix operand order repaired and checked by reading, not compiled, not run on hardware
//
#include <hip/hip_runtime.h>
#include <hip/hip_bf16.h>
#include <math.h>

// Problem constants (match reference)
#define BB 32
#define TT 2048
#define HH 1024

typedef __attribute__((ext_vector_type(16))) _Float16 v16h;
typedef __attribute__((ext_vector_type(8)))  _Float16 v8h;
typedef __attribute__((ext_vector_type(4)))  _Float16 v4h;
typedef __attribute__((ext_vector_type(8)))  float    v8f;

union AFrag { v16h v; v8h h[2]; };

// ---------------------------------------------------------------------------
// Kernel 1: transpose+convert W1 = W_hid[0:H, :] (f32, [k][n]) into
//           W1t (f16, [n][k]) so WMMA B-fragments are contiguous per lane.
// ---------------------------------------------------------------------------
__global__ __launch_bounds__(256) void k_pack_w1t(const float* __restrict__ Whid,
                                                  _Float16* __restrict__ W1t) {
    __shared__ float tile[32][33];
    const int n0 = blockIdx.x * 32;
    const int k0 = blockIdx.y * 32;
    const int tx = threadIdx.x;   // 0..31
    const int ty = threadIdx.y;   // 0..7
#pragma unroll
    for (int j = 0; j < 4; ++j) {
        int k = k0 + ty + j * 8;
        tile[ty + j * 8][tx] = Whid[(size_t)k * HH + (n0 + tx)];  // coalesced in n
    }
    __syncthreads();
#pragma unroll
    for (int j = 0; j < 4; ++j) {
        int n = n0 + ty + j * 8;
        W1t[(size_t)n * HH + (k0 + tx)] = (_Float16)tile[tx][ty + j * 8]; // coalesced in k
    }
}

// ---------------------------------------------------------------------------
// Kernel 2: u[b][h] = b_hid[h] + sum_f hidden[b][f] * W_hid[H+f][h]
// ---------------------------------------------------------------------------
__global__ __launch_bounds__(256) void k_compute_u(const float* __restrict__ hidden,
                                                   const float* __restrict__ Whid,
                                                   const float* __restrict__ bhid,
                                                   float* __restrict__ u) {
    const int b   = blockIdx.x;
    const int tid = threadIdx.x;
    float acc[4];
#pragma unroll
    for (int j = 0; j < 4; ++j) acc[j] = bhid[tid + j * 256];
    for (int f = 0; f < HH; ++f) {
        const float hv = hidden[b * HH + f];
        const float* row = Whid + (size_t)(HH + f) * HH;
#pragma unroll
        for (int j = 0; j < 4; ++j) acc[j] += hv * row[tid + j * 256];
    }
#pragma unroll
    for (int j = 0; j < 4; ++j) u[b * HH + tid + j * 256] = acc[j];
}

// ---------------------------------------------------------------------------
// Kernel 3: fused scores GEMM.
//   WG = 512 threads = 16 waves, owns (b, 64 rows of T)   [BM = 64]
//   Wave w: M-group (w>>3) -> M-tiles {2g, 2g+1};  N-range (w&7)*128, 8 N-tiles
//           -> 16 v8f accumulators per wave.
//   A panel (64 x 1024 f32 -> f16) staged in LDS in four K-chunks of 256.
//   B fragments software-pipelined in register arrays (4+4) so all 16
//   global_load_b128 of a k-step are in flight before the first WMMA waits.
//   Epilogue: tanh(c + u) * w_score, shfl+LDS reduction -> scores[b][t].
// ---------------------------------------------------------------------------
#define BM    64
#define KCH   256
#define ASTR  264   // padded f16 stride: 528 B/row -> LDS bank stride 4 (conflict-free b128)

__global__ __launch_bounds__(512) void k_scores(const float* __restrict__ E,
                                                const _Float16* __restrict__ W1t,
                                                const float* __restrict__ u,
                                                const float* __restrict__ wscore,
                                                float* __restrict__ scores) {
    __shared__ __align__(16) _Float16 Alds[BM * ASTR];
    __shared__ float s_scores[BM];

    const int tid  = threadIdx.x;
    const int lane = tid & 31;
    const int wave = tid >> 5;          // 0..15
    const int lm   = lane & 15;         // N-col within tile / M-row within half
    const int lh   = lane >> 4;         // 0: lanes 0-15, 1: lanes 16-31

    const int b  = blockIdx.x >> 5;           // T/64 = 32 row-blocks per batch
    const int m0 = (blockIdx.x & 31) * BM;

    const int mg  = wave >> 3;          // M-group 0/1 -> rows mg*32 .. mg*32+31
    const int n0w = (wave & 7) * 128;   // this wave's N-range

    if (tid < BM) s_scores[tid] = 0.0f;

    v8f c[2][8];
    {
        v8f z = {};
#pragma unroll
        for (int mi = 0; mi < 2; ++mi)
#pragma unroll
            for (int ni = 0; ni < 8; ++ni) c[mi][ni] = z;
    }

    const float* srcA = E + ((size_t)b * TT + m0) * HH;
    // per-lane base of this wave's B columns: column (n0w+lm), K-half lh
    const _Float16* bptr = W1t + (size_t)(n0w + lm) * HH + lh * 16;

    for (int kh = 0; kh < 4; ++kh) {
        // ---- stage A chunk: 64 rows x 256 cols, f32 -> f16 into LDS ----
#pragma unroll
        for (int i = 0; i < 8; ++i) {
            const int idx = tid + i * 512;      // 4096 float4 chunks
            const int row = idx >> 6;           // /64
            const int col = (idx & 63) * 4;
            const float* sp = srcA + (size_t)row * HH + kh * KCH + col;
            const float4 f = *reinterpret_cast<const float4*>(sp);
            if (kh < 3) __builtin_prefetch(sp + KCH, 0, 1);  // global_prefetch_b8 next chunk
            v4h h;
            h.x = (_Float16)f.x; h.y = (_Float16)f.y;
            h.z = (_Float16)f.z; h.w = (_Float16)f.w;
            *reinterpret_cast<v4h*>(&Alds[row * ASTR + col]) = h;
        }
        __syncthreads();

        // ---- compute: 8 K-steps of 32 over this chunk ----
        for (int ks = 0; ks < 8; ++ks) {
            const int kloc  = ks * 32;
            const int kglob = kh * KCH + kloc;

            // A fragments for both M-tiles (contiguous K layout matching B: lanes 0-15 K{0..15},
            // lanes 16-31 K{16..31})
            AFrag a[2];
#pragma unroll
            for (int mi = 0; mi < 2; ++mi) {
                const int base = (mg * 32 + mi * 16 + lm) * ASTR + kloc + lh * 16;
                a[mi].h[0] = *reinterpret_cast<const v8h*>(&Alds[base]);
                a[mi].h[1] = *reinterpret_cast<const v8h*>(&Alds[base + 8]);
            }

            // Preload ALL B fragments for this k-step first (16 b128 loads in
            // flight), then run the 16 WMMAs -- avoids waitcnt-0 stalls per tile.
            v16h b0[4], b1[4];
#pragma unroll
            for (int ni = 0; ni < 4; ++ni)
                b0[ni] = *reinterpret_cast<const v16h*>(bptr + (size_t)ni * 16 * HH + kglob);
#pragma unroll
            for (int ni = 0; ni < 4; ++ni)
                b1[ni] = *reinterpret_cast<const v16h*>(bptr + (size_t)(ni + 4) * 16 * HH + kglob);

#pragma unroll
            for (int ni = 0; ni < 4; ++ni) {
                c[0][ni] = __builtin_amdgcn_wmma_f32_16x16x32_f16(
                    false, a[0].v, false, b0[ni], (short)0, c[0][ni], false, false);
                c[1][ni] = __builtin_amdgcn_wmma_f32_16x16x32_f16(
                    false, a[1].v, false, b0[ni], (short)0, c[1][ni], false, false);
            }
#pragma unroll
            for (int ni = 0; ni < 4; ++ni) {
                c[0][ni + 4] = __builtin_amdgcn_wmma_f32_16x16x32_f16(
                    false, a[0].v, false, b1[ni], (short)0, c[0][ni + 4], false, false);
                c[1][ni + 4] = __builtin_amdgcn_wmma_f32_16x16x32_f16(
                    false, a[1].v, false, b1[ni], (short)0, c[1][ni + 4], false, false);
            }
        }
        __syncthreads();   // protect LDS before restage / before epilogue atomics
    }

    // ---- epilogue: tanh(c + u[b,n]) * w_score[n], reduce over n ----
#pragma unroll
    for (int mi = 0; mi < 2; ++mi) {
        float srow[8];
#pragma unroll
        for (int r = 0; r < 8; ++r) srow[r] = 0.0f;
#pragma unroll
        for (int ni = 0; ni < 8; ++ni) {
            const int n = n0w + ni * 16 + lm;
            const float uval = u[b * HH + n];
            const float wsv  = wscore[n];
#pragma unroll
            for (int r = 0; r < 8; ++r)
                srow[r] += tanhf(c[mi][ni][r] + uval) * wsv;
        }
#pragma unroll
        for (int r = 0; r < 8; ++r) {
            float v = srow[r];
            // reduce over the 16 lanes that share a row (C layout: M splits at lane 16)
            v += __shfl_xor(v, 8, 16);
            v += __shfl_xor(v, 4, 16);
            v += __shfl_xor(v, 2, 16);
            v += __shfl_xor(v, 1, 16);
            if (lm == 0)
                atomicAdd(&s_scores[mg * 32 + mi * 16 + lh * 8 + r], v); // ds_add_f32
        }
    }
    __syncthreads();
    if (tid < BM) scores[(size_t)b * TT + m0 + tid] = s_scores[tid];
}

// ---------------------------------------------------------------------------
// Kernel 4: softmax over T per batch; writes attn_weights into output.
// ---------------------------------------------------------------------------
__global__ __launch_bounds__(256) void k_softmax(const float* __restrict__ scores,
                                                 float* __restrict__ wout) {
    __shared__ float s[TT];
    __shared__ float red[256];
    const int b = blockIdx.x, tid = threadIdx.x;

    float m = -INFINITY;
    for (int i = tid; i < TT; i += 256) {
        const float v = scores[(size_t)b * TT + i];
        s[i] = v;
        m = fmaxf(m, v);
    }
    red[tid] = m; __syncthreads();
    for (int o = 128; o > 0; o >>= 1) {
        if (tid < o) red[tid] = fmaxf(red[tid], red[tid + o]);
        __syncthreads();
    }
    const float mx = red[0];
    __syncthreads();

    float sum = 0.0f;
    for (int i = tid; i < TT; i += 256) {
        const float e = expf(s[i] - mx);
        s[i] = e;
        sum += e;
    }
    red[tid] = sum; __syncthreads();
    for (int o = 128; o > 0; o >>= 1) {
        if (tid < o) red[tid] += red[tid + o];
        __syncthreads();
    }
    const float inv = 1.0f / red[0];
    for (int i = tid; i < TT; i += 256)
        wout[(size_t)b * TT + i] = s[i] * inv;
}

// ---------------------------------------------------------------------------
// Kernel 5: context[b][h] = sum_t w[b][t] * E[b][t][h]   (memory-bound pass)
// ---------------------------------------------------------------------------
__global__ __launch_bounds__(256) void k_context(const float* __restrict__ E,
                                                 const float* __restrict__ wts,
                                                 float* __restrict__ ctx) {
    __shared__ float w[TT];
    const int b = blockIdx.y;
    const int h = blockIdx.x * 256 + threadIdx.x;
    for (int i = threadIdx.x; i < TT; i += 256) w[i] = wts[(size_t)b * TT + i];
    __syncthreads();
    const float* e = E + (size_t)b * TT * HH + h;
    float acc = 0.0f;
#pragma unroll 4
    for (int t = 0; t < TT; ++t) acc += w[t] * e[(size_t)t * HH];
    ctx[b * HH + h] = acc;
}

// ---------------------------------------------------------------------------
extern "C" void kernel_launch(void* const* d_in, const int* in_sizes, int n_in,
                              void* d_out, int out_size, void* d_ws, size_t ws_size,
                              hipStream_t stream) {
    const float* hidden = (const float*)d_in[0];   // [B,1,H]
    const float* E      = (const float*)d_in[1];   // [B,T,H]
    const float* Whid   = (const float*)d_in[2];   // [2H,H]
    const float* bhid   = (const float*)d_in[3];   // [H]
    const float* wscore = (const float*)d_in[4];   // [H]
    float* out = (float*)d_out;                    // [B*H context | B*T weights]

    char* ws = (char*)d_ws;
    _Float16* W1t   = (_Float16*)ws;                               // 2 MB
    float*    u     = (float*)(ws + (size_t)HH * HH * 2);          // 128 KB
    float*    score = (float*)(ws + (size_t)HH * HH * 2 + BB * HH * 4); // 256 KB

    k_pack_w1t<<<dim3(HH / 32, HH / 32), dim3(32, 8), 0, stream>>>(Whid, W1t);
    k_compute_u<<<BB, 256, 0, stream>>>(hidden, Whid, bhid, u);
    k_scores<<<BB * (TT / BM), 512, 0, stream>>>(E, W1t, u, wscore, score);
    k_softmax<<<BB, 256, 0, stream>>>(score, out + BB * HH);
    k_context<<<dim3(HH / 256, BB), 256, 0, stream>>>(E, out + BB * HH, out);
}
